// LineGraphHead_17377437679650
// MI455X (gfx1250) — matrix operand order repaired and checked by reading, not compiled, not run on hardware
//
#include <hip/hip_runtime.h>
#include <hip/hip_bf16.h>
#include <stdint.h>

// MI455X / gfx1250: wave32, WMMA 16x16x32 bf16 (f32 accum).

typedef __attribute__((ext_vector_type(16))) __bf16 v16bf;
typedef __attribute__((ext_vector_type(8)))  float  v8f;

#define NN     50000
#define EDIR   800000
#define EHALF  400000
#define DDIM   256
#define GG     1000

// ---------------- helpers ----------------

__device__ __forceinline__ unsigned short f2bf(float f) {
  unsigned int u = __float_as_uint(f);
  return (unsigned short)((u + 0x7FFFu + ((u >> 16) & 1u)) >> 16);  // RNE
}

__device__ __forceinline__ void store_bf8(unsigned short* dst,
    float f0,float f1,float f2,float f3,float f4,float f5,float f6,float f7) {
  uint4 q;
  q.x = (unsigned)f2bf(f0) | ((unsigned)f2bf(f1) << 16);
  q.y = (unsigned)f2bf(f2) | ((unsigned)f2bf(f3) << 16);
  q.z = (unsigned)f2bf(f4) | ((unsigned)f2bf(f5) << 16);
  q.w = (unsigned)f2bf(f6) | ((unsigned)f2bf(f7) << 16);
  *(uint4*)dst = q;
}

// Fused GEMM layer on MT 16-row tiles (MT*16 rows total):
//   out[MT*16][Nt*16] = relu( actIn[MT*16][Kt*32] (bf16) @ Wpacked + bias )
// A fragment per CDNA5 ISA 16-bit A 16x32 layout:
//   lane (half=l>>4, m=l&15): elems j<8 -> K = kbase+8*half+j ; j>=8 -> K = kbase+16+8*half+(j-8)
// B fragment pre-swizzled in global memory: lane-contiguous 16 bf16 per (kt,nt,lane);
// each B fragment is reused by MT WMMAs (halves L2 weight traffic for MT=2).
// C/D: VGPR r -> row M = r + 8*half, col N = m.
template<int MT, bool WBF>
__device__ __forceinline__ void gemm_layer(
    const unsigned short* actIn, int strideIn,
    const unsigned short* __restrict__ wp, const float* __restrict__ bias,
    int Kt, int Nt,
    unsigned short* outBf, float* outF, int strideOut, int lane)
{
  const int half = lane >> 4, m = lane & 15;
  for (int nt = 0; nt < Nt; nt++) {
    const float bv = bias[nt * 16 + m];   // all 8 C elems of a lane share column N = nt*16+m
    v8f acc[MT];
    #pragma unroll
    for (int mt = 0; mt < MT; mt++)
      #pragma unroll
      for (int r = 0; r < 8; r++) acc[mt][r] = bv;

    #pragma unroll 4
    for (int kt = 0; kt < Kt; kt++) {
      union { v16bf v; uint4 q[2]; } B;
      const unsigned short* bp = wp + ((size_t)(kt * Nt + nt) * 32 + lane) * 16;
      B.q[0] = *(const uint4*)(bp);       // coalesced global_load_b128, L2-hot
      B.q[1] = *(const uint4*)(bp + 8);
      #pragma unroll
      for (int mt = 0; mt < MT; mt++) {
        union { v16bf v; uint4 q[2]; } A;
        const unsigned short* ar = actIn + (mt * 16 + m) * strideIn + kt * 32 + 8 * half;
        A.q[0] = *(const uint4*)(ar);     // ds_load_b128: K = kbase+8h..+7
        A.q[1] = *(const uint4*)(ar + 16);// ds_load_b128: K = kbase+16+8h..+7
        acc[mt] = __builtin_amdgcn_wmma_f32_16x16x32_bf16(
            false, A.v, false, B.v, (short)0, acc[mt], false, false);
      }
    }

    #pragma unroll
    for (int mt = 0; mt < MT; mt++) {
      #pragma unroll
      for (int r = 0; r < 8; r++) {
        float vv = fmaxf(acc[mt][r], 0.0f);       // ReLU
        int row = mt * 16 + r + 8 * half;
        int col = nt * 16 + m;
        if (WBF) outBf[row * strideOut + col] = f2bf(vv);   // compile-time selected
        else     outF [row * strideOut + col] = vv;
      }
    }
  }
}

// Segment-sum pool of a ROWS x 64 f32 tile into accG[G][64].
// batch is sorted -> run-length flush keeps atomics ~1 per column per tile.
template<int ROWS>
__device__ __forceinline__ void pool_tile(const float* outBuf, int strideOut,
    const int* __restrict__ batch, int base, float* __restrict__ accG, int lane)
{
  for (int cc = 0; cc < 2; cc++) {
    int c = lane + cc * 32;
    int curg = batch[base];
    float run = 0.0f;
    #pragma unroll 4
    for (int row = 0; row < ROWS; row++) {
      int g2 = batch[base + row];
      if (g2 != curg) {
        atomicAdd(&accG[(size_t)curg * 64 + c], run);
        run = 0.0f; curg = g2;
      }
      run += outBuf[row * strideOut + c];
    }
    atomicAdd(&accG[(size_t)curg * 64 + c], run);
  }
}

// ---------------- prep kernels ----------------

__global__ void zero_f32(float* p, int n) {
  int i = blockIdx.x * 256 + threadIdx.x;
  if (i < n) p[i] = 0.0f;
}

// W[K][N] fp32 row-major -> packed bf16 B-fragments:
// P[((kt*Nt+nt)*32 + lane)*16 + j] = bf16( W[kt*32 + 8*(lane>>4) + j + (j>=8 ? 8 : 0)][nt*16 + (lane&15)] )
__global__ void pack_weights(const float* __restrict__ W, unsigned short* __restrict__ P,
                             int K, int Nn) {
  int idx = blockIdx.x * 256 + threadIdx.x;
  if (idx >= K * Nn) return;
  int j    = idx & 15;
  int lane = (idx >> 4) & 31;
  int t    = idx >> 9;
  int Nt = Nn >> 4;
  int kt = t / Nt, nt = t - kt * Nt;
  int half = lane >> 4, m = lane & 15;
  int k = kt * 32 + 8 * half + j + (j >= 8 ? 8 : 0);
  int n = nt * 16 + m;
  P[idx] = f2bf(W[(size_t)k * Nn + n]);
}

__global__ void count_kernel(const int* __restrict__ batch, float* __restrict__ cnt, int n) {
  int i = blockIdx.x * 256 + threadIdx.x;
  if (i < n) atomicAdd(&cnt[batch[i]], 1.0f);
}

// ---------------- node MLP: x -> 128 -> 64, pooled (16 nodes / wave) ----------------

__global__ void __launch_bounds__(32)
node_mlp_kernel(const float* __restrict__ x, const int* __restrict__ node_batch,
                const unsigned short* __restrict__ pw0, const unsigned short* __restrict__ pw1,
                const float* __restrict__ b0, const float* __restrict__ b1,
                float* __restrict__ nodeAcc)
{
  __shared__ alignas(16) unsigned short sIN[16 * 264];    // K=256, padded stride
  __shared__ alignas(16) unsigned short sACT1[16 * 136];  // K=128
  const int lane = threadIdx.x;
  const int base = blockIdx.x * 16;
  const int c0 = lane * 8;

  for (int row = 0; row < 16; row++) {
    const float* xr = x + (size_t)(base + row) * DDIM + c0;
    float4 a = *(const float4*)(xr);
    float4 b = *(const float4*)(xr + 4);
    store_bf8(&sIN[row * 264 + c0], a.x, a.y, a.z, a.w, b.x, b.y, b.z, b.w);
  }
  __syncthreads();
  gemm_layer<1, true >(sIN, 264, pw0, b0, /*Kt=*/8, /*Nt=*/8, sACT1, nullptr, 136, lane);
  __syncthreads();
  float* outBuf = (float*)sIN;                            // reuse: layer0 input dead
  gemm_layer<1, false>(sACT1, 136, pw1, b1, /*Kt=*/4, /*Nt=*/4, nullptr, outBuf, 68, lane);
  __syncthreads();
  pool_tile<16>(outBuf, 68, node_batch, base, nodeAcc, lane);
}

// ---------------- edge MLP: [e | x[src]+x[dst]] (512) -> 256 -> 128 -> 64, pooled ----------------
// 32 edges (two M-tiles) per wave: every B fragment feeds two WMMAs.

__global__ void __launch_bounds__(32)
edge_mlp_kernel(const float* __restrict__ e, const float* __restrict__ x,
                const int* __restrict__ ei, const int* __restrict__ e_batch,
                const unsigned short* __restrict__ pw0, const unsigned short* __restrict__ pw1,
                const unsigned short* __restrict__ pw2,
                const float* __restrict__ b0, const float* __restrict__ b1,
                const float* __restrict__ b2,
                float* __restrict__ edgeAcc)
{
  __shared__ alignas(16) unsigned short sIN[32 * 520];    // K=512 input tile (33280 B)
  __shared__ alignas(16) unsigned short sACT1[32 * 264];  // K=256 (16896 B)
  const int lane = threadIdx.x;
  const int base = blockIdx.x * 32;
  const int c0 = lane * 8;

  for (int row = 0; row < 32; row++) {
    int eidx = base + row;
    const float* er = e + (size_t)eidx * DDIM + c0;
    float4 a = *(const float4*)(er);
    float4 b = *(const float4*)(er + 4);
    store_bf8(&sIN[row * 520 + c0], a.x, a.y, a.z, a.w, b.x, b.y, b.z, b.w);
    int s = ei[eidx];               // first EHALF columns have src < dst (== reference idx)
    int d = ei[EDIR + eidx];
    const float* xs = x + (size_t)s * DDIM + c0;
    const float* xd = x + (size_t)d * DDIM + c0;
    float4 s0 = *(const float4*)(xs);
    float4 s1 = *(const float4*)(xs + 4);
    float4 d0 = *(const float4*)(xd);
    float4 d1 = *(const float4*)(xd + 4);
    store_bf8(&sIN[row * 520 + 256 + c0],
              s0.x + d0.x, s0.y + d0.y, s0.z + d0.z, s0.w + d0.w,
              s1.x + d1.x, s1.y + d1.y, s1.z + d1.z, s1.w + d1.w);
  }
  __syncthreads();
  gemm_layer<2, true >(sIN, 520, pw0, b0, /*Kt=*/16, /*Nt=*/16, sACT1, nullptr, 264, lane);
  __syncthreads();
  unsigned short* sACT2 = sIN;                            // alias: [32][136] = 8704 B
  gemm_layer<2, true >(sACT1, 264, pw1, b1, /*Kt=*/8, /*Nt=*/8, sACT2, nullptr, 136, lane);
  __syncthreads();
  float* outBuf = (float*)((char*)sIN + 8704);            // [32][68] f32, disjoint from sACT2
  gemm_layer<2, false>(sACT2, 136, pw2, b2, /*Kt=*/4, /*Nt=*/4, nullptr, outBuf, 68, lane);
  __syncthreads();
  pool_tile<32>(outBuf, 68, e_batch, base, edgeAcc, lane);
}

// ---------------- final: mean, concat, [G,128]@[128,1]+ob ; emit pred then y ----------------

__global__ void final_kernel(const float* __restrict__ nodeAcc, const float* __restrict__ edgeAcc,
                             const float* __restrict__ nodeCnt, const float* __restrict__ edgeCnt,
                             const float* __restrict__ ow, const float* __restrict__ ob,
                             const float* __restrict__ y, float* __restrict__ out)
{
  int g = blockIdx.x * 256 + threadIdx.x;
  if (g >= GG) return;
  float ninv = 1.0f / fmaxf(nodeCnt[g], 1.0f);
  float einv = 1.0f / fmaxf(edgeCnt[g], 1.0f);
  float s = ob[0];
  #pragma unroll 8
  for (int c = 0; c < 64; c++) s += nodeAcc[(size_t)g * 64 + c] * ninv * ow[c];
  #pragma unroll 8
  for (int c = 0; c < 64; c++) s += edgeAcc[(size_t)g * 64 + c] * einv * ow[64 + c];
  out[g] = s;
  out[GG + g] = y[g];
}

// ---------------- host ----------------

extern "C" void kernel_launch(void* const* d_in, const int* in_sizes, int n_in,
                              void* d_out, int out_size, void* d_ws, size_t ws_size,
                              hipStream_t stream) {
  (void)in_sizes; (void)n_in; (void)out_size; (void)ws_size;
  const float* x     = (const float*)d_in[0];
  const float* e     = (const float*)d_in[1];
  const int*   ei    = (const int*)  d_in[2];
  const int*   nb    = (const int*)  d_in[3];
  const int*   ebat  = (const int*)  d_in[4];
  const float* y     = (const float*)d_in[5];
  const float* fc_w0 = (const float*)d_in[6];
  const float* fc_b0 = (const float*)d_in[7];
  const float* fc_w1 = (const float*)d_in[8];
  const float* fc_b1 = (const float*)d_in[9];
  const float* ew0   = (const float*)d_in[10];
  const float* eb0   = (const float*)d_in[11];
  const float* ew1   = (const float*)d_in[12];
  const float* eb1   = (const float*)d_in[13];
  const float* ew2   = (const float*)d_in[14];
  const float* eb2   = (const float*)d_in[15];
  const float* ow    = (const float*)d_in[16];
  const float* ob    = (const float*)d_in[17];

  char* ws = (char*)d_ws;
  float* nodeAcc = (float*)(ws + 0);            // G*64 f32
  float* edgeAcc = (float*)(ws + 256000);       // G*64 f32
  float* nodeCnt = (float*)(ws + 512000);       // G f32
  float* edgeCnt = (float*)(ws + 516000);       // G f32
  unsigned short* pw_fc0 = (unsigned short*)(ws + 520000);  // 256x128 bf16
  unsigned short* pw_fc1 = (unsigned short*)(ws + 585536);  // 128x64
  unsigned short* pw_e0  = (unsigned short*)(ws + 601920);  // 512x256
  unsigned short* pw_e1  = (unsigned short*)(ws + 864064);  // 256x128
  unsigned short* pw_e2  = (unsigned short*)(ws + 929600);  // 128x64   (total ~0.92 MB)

  // zero accumulators + counts (contiguous 130000 floats)
  zero_f32<<<(130000 + 255) / 256, 256, 0, stream>>>(nodeAcc, 130000);

  pack_weights<<<(256 * 128 + 255) / 256, 256, 0, stream>>>(fc_w0, pw_fc0, 256, 128);
  pack_weights<<<(128 *  64 + 255) / 256, 256, 0, stream>>>(fc_w1, pw_fc1, 128,  64);
  pack_weights<<<(512 * 256 + 255) / 256, 256, 0, stream>>>(ew0,   pw_e0,  512, 256);
  pack_weights<<<(256 * 128 + 255) / 256, 256, 0, stream>>>(ew1,   pw_e1,  256, 128);
  pack_weights<<<(128 *  64 + 255) / 256, 256, 0, stream>>>(ew2,   pw_e2,  128,  64);

  count_kernel<<<(NN    + 255) / 256, 256, 0, stream>>>(nb,   nodeCnt, NN);
  count_kernel<<<(EHALF + 255) / 256, 256, 0, stream>>>(ebat, edgeCnt, EHALF);

  node_mlp_kernel<<<NN / 16, 32, 0, stream>>>(x, nb, pw_fc0, pw_fc1, fc_b0, fc_b1, nodeAcc);
  edge_mlp_kernel<<<EHALF / 32, 32, 0, stream>>>(e, x, ei, ebat, pw_e0, pw_e1, pw_e2,
                                                 eb0, eb1, eb2, edgeAcc);

  final_kernel<<<(GG + 255) / 256, 256, 0, stream>>>(nodeAcc, edgeAcc, nodeCnt, edgeCnt,
                                                     ow, ob, y, (float*)d_out);
}
